// UResNet_87308095193430
// MI455X (gfx1250) — compile-verified, matrix operand order checked by reading, not run on hardware
//
#include <hip/hip_runtime.h>

// ---------------------------------------------------------------------------
// Sparse 3D U-Net forward for MI455X (gfx1250, wave32).
// Convs are implicit-GEMM on v_wmma_f32_16x16x32_f16 (f16 operands, f32 acc).
//  - K-chunks are tap-uniform: A fragment = 4x global_load_b128 + cvt_pk.
//  - (tap,cin) decode is incremental uniform SALU (no divisions in the loop).
//  - B fragments pre-packed to WMMA lane layout (f16) by k_packB: 1x 32B load
//    + global_prefetch of the next chunk.
//  - BN-ReLU fused into the conv A-read (scale/shift per channel + input mask),
//    final BN fused into the gather. No materialized bnrelu tensors.
// All tensors NDHWC fp32, zero outside the occupancy mask (invariant).
// ---------------------------------------------------------------------------

typedef __attribute__((ext_vector_type(16))) _Float16 v16h;
typedef __attribute__((ext_vector_type(8)))  float    v8f;

#define MODE_CONV3 0   // 3x3x3 SAME conv, same resolution
#define MODE_DOWN  1   // 2x2x2 stride-2 VALID conv (out = half res)
#define MODE_UP    2   // 2x2x2 stride-2 transpose conv (out = double res)
#define MODE_NIN   3   // 1x1x1 conv (channel GEMM)

// Pack weight matrix [Ktot x Cout] into WMMA-B lane layout, f16, zero-padded.
// dst element block: ((ck*nG + g)*32 + lane)*16 halves, 32B per lane-fragment.
__global__ void k_packB(const float* __restrict__ w, int Ktot, int Cout,
                        int nG, int nChunks, _Float16* __restrict__ dst)
{
    const int tid = blockIdx.x * blockDim.x + threadIdx.x;
    const int total = nChunks * nG * 32;
    if (tid >= total) return;
    const int lane = tid & 31;
    const int g    = (tid >> 5) % nG;
    const int ck   = (tid >> 5) / nG;
    const int kb   = (lane >> 4) * 8;
    const int n    = g * 16 + (lane & 15);
    const int k0   = ck << 5;
    v16h vals;
    #pragma unroll
    for (int r = 0; r < 8; ++r) {
        const int Kr = (r < 4 ? 2 * r : 16 + 2 * (r - 4)) + kb;
        #pragma unroll
        for (int j = 0; j < 2; ++j) {
            const int kg = k0 + Kr + j;
            float v = 0.f;
            if (kg < Ktot && n < Cout) v = w[(long)kg * Cout + n];
            vals[2 * r + j] = (_Float16)v;
        }
    }
    *(v16h*)(dst + (long)tid * 16) = vals;
}

// One wave per block. Tile = 16 output voxels x 16 output channels.
template <int MODE, bool BN>
__global__ __launch_bounds__(32)
void k_conv(const float* __restrict__ in, int Cin, int S_in,
            const _Float16* __restrict__ pB,
            float* out, int Cout, int outStride,
            const float* __restrict__ maskOut,
            const float* __restrict__ maskIn,
            const float* __restrict__ bnScale, const float* __restrict__ bnShift,
            const float* resid, int S_out, int packPerTap)
{
    const int lane  = threadIdx.x;           // 0..31 (wave32)
    const int tile  = blockIdx.x;

    __shared__ float s_mask[16];
    __shared__ int   s_ovox[16];
    __shared__ int   s_d[16], s_h[16], s_w[16];

    const int Vout = S_out * S_out * S_out;

    if (lane < 16) {
        int tv = tile * 16 + lane;
        int ov = -1, id = 0, ih = 0, iw = 0;
        float mv = 0.f;
        if (MODE == MODE_UP) {
            int Vin = S_in * S_in * S_in;
            if (tv < Vin) {
                int dc = tv / (S_in * S_in);
                int hc = (tv / S_in) % S_in;
                int wc = tv % S_in;
                int tz = (blockIdx.z >> 2) & 1, ty = (blockIdx.z >> 1) & 1, tx = blockIdx.z & 1;
                int od = 2 * dc + tz, oh = 2 * hc + ty, ow = 2 * wc + tx;
                ov = (od * S_out + oh) * S_out + ow;
                id = dc; ih = hc; iw = wc;
                mv = maskOut[ov];
            }
        } else {
            if (tv < Vout) {
                ov = tv;
                int d = tv / (S_out * S_out);
                int h = (tv / S_out) % S_out;
                int x = tv % S_out;
                if (MODE == MODE_DOWN) { id = 2 * d; ih = 2 * h; iw = 2 * x; }
                else                   { id = d;     ih = h;     iw = x;     }
                mv = maskOut[ov];
            }
        }
        s_ovox[lane] = ov; s_mask[lane] = mv;
        s_d[lane] = id; s_h[lane] = ih; s_w[lane] = iw;
    }
    __syncthreads();

    const int anyActive = __any(s_mask[lane & 15] != 0.f);

    const int Mrow = lane & 15;
    const int kb   = (lane >> 4) * 8;
    // hoist loop-invariant per-lane tile data out of the K loop
    const int ovM = s_ovox[Mrow];
    const int bd0 = s_d[Mrow], bh0 = s_h[Mrow], bw0 = s_w[Mrow];

    const int taps = (MODE == MODE_CONV3) ? 27 : (MODE == MODE_DOWN) ? 8 : 1;
    const int Ktot = taps * Cin;
    const int nG   = gridDim.y;
    const int nChunks = (Ktot + 31) >> 5;
    const long chStride = (long)nG * 512;
    const _Float16* pbLane = pB
        + ((MODE == MODE_UP) ? (long)blockIdx.z * packPerTap : 0)
        + ((long)blockIdx.y * 32 + lane) * 16;

    v8f acc = {};
    if (anyActive) {
        if (MODE == MODE_CONV3 && Cin < 16) {
            // scalar fallback (conv_in, Cin==1); no BN fusion on this path
            for (int ck = 0; ck < nChunks; ++ck) {
                const v16h b = *(const v16h*)(pbLane + (long)ck * chStride);
                const int k0 = ck << 5;
                v16h a;
                #pragma unroll
                for (int r = 0; r < 8; ++r) {
                    const int Kr = (r < 4 ? 2 * r : 16 + 2 * (r - 4)) + kb;
                    #pragma unroll
                    for (int j = 0; j < 2; ++j) {
                        const int kg = k0 + Kr + j;
                        float av = 0.f;
                        if (kg < Ktot && ovM >= 0) {
                            const int tap = kg;              // Cin == 1
                            const int id = bd0 + tap / 9 - 1;
                            const int ih = bh0 + (tap / 3) % 3 - 1;
                            const int iw = bw0 + tap % 3 - 1;
                            if ((unsigned)id < (unsigned)S_in &&
                                (unsigned)ih < (unsigned)S_in &&
                                (unsigned)iw < (unsigned)S_in)
                                av = in[(id * S_in + ih) * (long)S_in + iw];
                        }
                        a[2 * r + j] = (_Float16)av;
                    }
                }
                acc = __builtin_amdgcn_wmma_f32_16x16x32_f16(
                          false, a, false, b, (short)0, acc, false, false);
            }
        } else {
            // incremental uniform (tap, cin) walk: no divisions in the loop
            int cinL = 0, tapL = 0, kxL = 0, kyL = 0, kzL = 0;
            for (int ck = 0; ck < nChunks; ++ck) {
                const v16h b = *(const v16h*)(pbLane + (long)ck * chStride);
                if (ck + 1 < nChunks)
                    __builtin_prefetch(pbLane + (long)(ck + 1) * chStride, 0, 0);
                const int k0 = ck << 5;
                // hi-half tap = lo-half tap, stepped once if cin wraps
                int cinH = cinL + 16, tapH = tapL, kxH = kxL, kyH = kyL, kzH = kzL;
                if (cinH >= Cin) {
                    cinH -= Cin; ++tapH;
                    if (MODE == MODE_CONV3) {
                        if (++kxH == 3) { kxH = 0; if (++kyH == 3) { kyH = 0; ++kzH; } }
                    }
                }
                const bool hiValid = (k0 + 16) < Ktot;

                v16h a;
                #pragma unroll
                for (int h = 0; h < 2; ++h) {
                    const bool valid = (h == 0) ? true : hiValid;
                    const int cin = (h == 0) ? cinL : cinH;
                    const int tp  = (h == 0) ? tapL : tapH;
                    int id = bd0, ih = bh0, iw = bw0;
                    bool ok = (ovM >= 0);
                    if (MODE == MODE_CONV3) {
                        const int kx = (h == 0) ? kxL : kxH;
                        const int ky = (h == 0) ? kyL : kyH;
                        const int kz = (h == 0) ? kzL : kzH;
                        id += kz - 1; ih += ky - 1; iw += kx - 1;
                        ok = ok && (unsigned)id < (unsigned)S_in
                                && (unsigned)ih < (unsigned)S_in
                                && (unsigned)iw < (unsigned)S_in;
                    } else if (MODE == MODE_DOWN) {
                        id += (tp >> 2) & 1; ih += (tp >> 1) & 1; iw += tp & 1;
                    }
                    const int co = cin + kb;
                    float4 x0 = {0, 0, 0, 0}, x1 = {0, 0, 0, 0};
                    float gate = 0.f;
                    if (valid && ok) {
                        const int vin = (id * S_in + ih) * S_in + iw;
                        const float mIn = BN ? maskIn[vin] : 1.f;
                        if (mIn != 0.f) {
                            const float* p = in + (long)vin * Cin + co;
                            x0 = *(const float4*)(p);
                            x1 = *(const float4*)(p + 4);
                            gate = 1.f;
                        }
                    }
                    float v[8] = {x0.x, x0.y, x0.z, x0.w, x1.x, x1.y, x1.z, x1.w};
                    if (BN) {
                        if (valid) {
                            const float4 sc0 = *(const float4*)(bnScale + co);
                            const float4 sc1 = *(const float4*)(bnScale + co + 4);
                            const float4 sh0 = *(const float4*)(bnShift + co);
                            const float4 sh1 = *(const float4*)(bnShift + co + 4);
                            const float sc[8] = {sc0.x, sc0.y, sc0.z, sc0.w,
                                                 sc1.x, sc1.y, sc1.z, sc1.w};
                            const float sh[8] = {sh0.x, sh0.y, sh0.z, sh0.w,
                                                 sh1.x, sh1.y, sh1.z, sh1.w};
                            #pragma unroll
                            for (int e = 0; e < 8; ++e)
                                v[e] = fmaxf(v[e] * sc[e] + sh[e], 0.f) * gate;
                        }
                    }
                    #pragma unroll
                    for (int e = 0; e < 8; ++e)
                        a[8 * h + e] = (_Float16)v[e];
                }
                acc = __builtin_amdgcn_wmma_f32_16x16x32_f16(
                          false, a, false, b, (short)0, acc, false, false);
                // advance lo-half state by 32 K
                cinL += 32;
                while (cinL >= Cin) {
                    cinL -= Cin; ++tapL;
                    if (MODE == MODE_CONV3) {
                        if (++kxL == 3) { kxL = 0; if (++kyL == 3) { kyL = 0; ++kzL; } }
                    }
                }
            }
        }
    }

    // C/D layout: lane l, reg r -> (M = (l>>4)*8 + r, N = l&15)
    const int n = lane & 15;
    const int cout0 = blockIdx.y * 16;
    if (cout0 + n < Cout) {
        #pragma unroll
        for (int r = 0; r < 8; ++r) {
            const int M  = ((lane >> 4) << 3) + r;
            const int ov = s_ovox[M];
            if (ov >= 0) {
                const long idx = (long)ov * outStride + cout0 + n;
                float v = acc[r] * s_mask[M];
                if (resid) v += resid[idx];
                out[idx] = v;
            }
        }
    }
}

// ---- BN stats: coalesced, fixed channel per thread (blockDim % C == 0). ----
__global__ void k_bnstats(const float* __restrict__ x, int V, int C,
                          float* __restrict__ sums)
{
    const int c = threadIdx.x % C;           // fixed: stride is multiple of C
    float s = 0.f, q = 0.f;
    const long tot = (long)V * C;
    for (long f = (long)blockIdx.x * blockDim.x + threadIdx.x; f < tot;
         f += (long)gridDim.x * blockDim.x) {
        const float val = x[f];
        s += val; q += val * val;
    }
    __shared__ float rs[256], rq[256];
    rs[threadIdx.x] = s; rq[threadIdx.x] = q;
    __syncthreads();
    for (int o = blockDim.x >> 1; o >= C; o >>= 1) {
        if ((int)threadIdx.x < o) {
            rs[threadIdx.x] += rs[threadIdx.x + o];
            rq[threadIdx.x] += rq[threadIdx.x + o];
        }
        __syncthreads();
    }
    if ((int)threadIdx.x < C) {
        atomicAdd(&sums[c], rs[threadIdx.x]);
        atomicAdd(&sums[256 + c], rq[threadIdx.x]);
    }
}

// sums -> per-channel scale/shift: y = max(x*scale + shift, 0)
__global__ void k_bnfin(const float* __restrict__ sums, const float* __restrict__ ncnt,
                        int C, float* __restrict__ scale, float* __restrict__ shift)
{
    const int c = threadIdx.x;
    if (c >= C) return;
    const float n    = *ncnt;
    const float mean = sums[c] / n;
    const float var  = sums[256 + c] / n - mean * mean;
    const float sc   = rsqrtf(var + 1e-4f);
    scale[c] = sc;
    shift[c] = -mean * sc;
}

__global__ void k_scatter(const int* __restrict__ coords, const float* __restrict__ feats,
                          int N, float* grid, float* mask)
{
    const int i = blockIdx.x * blockDim.x + threadIdx.x;
    if (i >= N) return;
    const int cx = coords[i * 3 + 0], cy = coords[i * 3 + 1], cz = coords[i * 3 + 2];
    const int v = (cx * 96 + cy) * 96 + cz;
    atomicAdd(&grid[v], feats[i]);
    mask[v] = 1.0f;
}

__global__ void k_maxpool(const float* __restrict__ mi, int So, float* __restrict__ mo)
{
    const int v = blockIdx.x * blockDim.x + threadIdx.x;
    const int Vo = So * So * So;
    if (v >= Vo) return;
    const int d = v / (So * So), h = (v / So) % So, w = v % So;
    const int Si = So * 2;
    float m = 0.f;
    #pragma unroll
    for (int a = 0; a < 2; ++a)
        #pragma unroll
        for (int b = 0; b < 2; ++b)
            #pragma unroll
            for (int c = 0; c < 2; ++c)
                m = fmaxf(m, mi[((2 * d + a) * Si + (2 * h + b)) * Si + (2 * w + c)]);
    mo[v] = m;
}

__global__ void k_count(const float* __restrict__ m, int V, float* out)
{
    float s = 0.f;
    for (int v = blockIdx.x * blockDim.x + threadIdx.x; v < V;
         v += gridDim.x * blockDim.x) s += m[v];
    __shared__ float r[256];
    r[threadIdx.x] = s; __syncthreads();
    for (int o = 128; o > 0; o >>= 1) {
        if ((int)threadIdx.x < o) r[threadIdx.x] += r[threadIdx.x + o];
        __syncthreads();
    }
    if (threadIdx.x == 0) atomicAdd(out, r[0]);
}

__global__ void k_copych(const float* __restrict__ x, float* __restrict__ cc,
                         unsigned V, int p)
{
    const unsigned f = blockIdx.x * blockDim.x + threadIdx.x;
    if (f >= V * (unsigned)p) return;
    const unsigned v = f / (unsigned)p;
    const unsigned c = f % (unsigned)p;
    cc[(long)v * 2 * p + c] = x[f];
}

// gather + fused final BN-ReLU + linear head (mask==1 at all gathered voxels)
__global__ void k_gather_lin(const float* __restrict__ x,
                             const int* __restrict__ coords, int N,
                             const float* __restrict__ sc, const float* __restrict__ sh,
                             const float* __restrict__ lw, const float* __restrict__ lb,
                             float* __restrict__ out)
{
    const int i = blockIdx.x * blockDim.x + threadIdx.x;
    if (i >= N) return;
    const int cx = coords[i * 3 + 0], cy = coords[i * 3 + 1], cz = coords[i * 3 + 2];
    const long v = (long)(cx * 96 + cy) * 96 + cz;
    float f[16];
    #pragma unroll
    for (int c = 0; c < 16; ++c)
        f[c] = fmaxf(x[v * 16 + c] * sc[c] + sh[c], 0.f);
    #pragma unroll
    for (int j = 0; j < 5; ++j) {
        float s = lb[j];
        #pragma unroll
        for (int c = 0; c < 16; ++c) s += f[c] * lw[c * 5 + j];
        out[i * 5 + j] = s;
    }
}

// ---------------------------------------------------------------------------

extern "C" void kernel_launch(void* const* d_in, const int* in_sizes, int n_in,
                              void* d_out, int out_size, void* d_ws, size_t ws_size,
                              hipStream_t stream)
{
    (void)n_in; (void)out_size; (void)ws_size;
    const int N = in_sizes[0] / 3;

    const int*   coords    = (const int*)d_in[0];
    const float* features  = (const float*)d_in[1];
    const float* conv_in_w = (const float*)d_in[2];

    struct LvlW { const float *q0w1,*q0w2,*q1w1,*q1w2,*down,*up,*nin,*p0w1,*p0w2,*p1w1,*p1w2; };
    LvlW Lw[5] = {};

    // Detect param flatten order from sizes (alphabetical vs insertion).
    const bool alpha = (in_sizes[3] == 4096);
    int idx = 3;
    for (int i = 0; i < 4; ++i) {
        const float* p[11];
        for (int j = 0; j < 11; ++j) p[j] = (const float*)d_in[idx + j];
        if (alpha) {
            Lw[i].down = p[0];  Lw[i].nin  = p[1];
            Lw[i].p0w1 = p[2];  Lw[i].p0w2 = p[3];
            Lw[i].p1w1 = p[4];  Lw[i].p1w2 = p[5];
            Lw[i].q0w1 = p[6];  Lw[i].q0w2 = p[7];
            Lw[i].q1w1 = p[8];  Lw[i].q1w2 = p[9];
            Lw[i].up   = p[10];
        } else {
            Lw[i].q0w1 = p[0];  Lw[i].q0w2 = p[1];
            Lw[i].q1w1 = p[2];  Lw[i].q1w2 = p[3];
            Lw[i].down = p[4];  Lw[i].up   = p[5];
            Lw[i].nin  = p[6];
            Lw[i].p0w1 = p[7];  Lw[i].p0w2 = p[8];
            Lw[i].p1w1 = p[9];  Lw[i].p1w2 = p[10];
        }
        idx += 11;
    }
    Lw[4].q0w1 = (const float*)d_in[idx + 0];
    Lw[4].q0w2 = (const float*)d_in[idx + 1];
    Lw[4].q1w1 = (const float*)d_in[idx + 2];
    Lw[4].q1w2 = (const float*)d_in[idx + 3];
    idx += 4;
    const float *lin_w, *lin_b;
    if (in_sizes[idx] == 5) { lin_b = (const float*)d_in[idx]; lin_w = (const float*)d_in[idx + 1]; }
    else                    { lin_w = (const float*)d_in[idx]; lin_b = (const float*)d_in[idx + 1]; }

    // ---- workspace carve-out (32B aligned blocks) ----
    const int  Sl[5] = {96, 48, 24, 12, 6};
    const int  pl[5] = {16, 32, 48, 64, 80};
    long Vl[5];
    for (int l = 0; l < 5; ++l) Vl[l] = (long)Sl[l] * Sl[l] * Sl[l];

    float* W = (float*)d_ws;
    long off = 0;
    auto alloc = [&](long nf) { nf = (nf + 7) & ~7L; float* p = W + off; off += nf; return p; };
    float* grid0 = alloc(Vl[0]);
    float* mask[5];
    for (int l = 0; l < 5; ++l) mask[l] = alloc(Vl[l]);
    float* x[5];
    for (int l = 0; l < 5; ++l) x[l] = alloc(Vl[l] * pl[l]);
    float* T2 = alloc(Vl[0] * 16);
    float* CC = alloc(Vl[0] * 32);
    float* bns      = alloc(512);
    float* ncnt     = alloc(8);
    float* scaleBuf = alloc(128);
    float* shiftBuf = alloc(128);
    _Float16* packBuf = (_Float16*)alloc(262144);   // 512K halves

    auto bnprep = [&](const float* xin, long V, int C, int lvl) {
        hipMemsetAsync(bns, 0, 512 * sizeof(float), stream);
        const int bd = (C % 48 == 0) ? 192 : (C == 80 ? 160 : 256);
        k_bnstats<<<256, bd, 0, stream>>>(xin, (int)V, C, bns);
        k_bnfin<<<1, 128, 0, stream>>>(bns, ncnt + lvl, C, scaleBuf, shiftBuf);
    };
    auto pack = [&](const float* w, int Ktot, int Cout, _Float16* dst) {
        const int nCh = (Ktot + 31) / 32, nG = (Cout + 15) / 16;
        const int tot = nCh * nG * 32;
        k_packB<<<(tot + 255) / 256, 256, 0, stream>>>(w, Ktot, Cout, nG, nCh, dst);
    };
    auto conv3 = [&](const float* in, int Cin, int S, const float* w, float* outp,
                     int Cout, const float* mk, bool bnfuse, const float* resid) {
        pack(w, 27 * Cin, Cout, packBuf);
        const long Vo = (long)S * S * S;
        dim3 g((unsigned)((Vo + 15) / 16), (unsigned)((Cout + 15) / 16));
        if (bnfuse)
            k_conv<MODE_CONV3, true><<<g, 32, 0, stream>>>(
                in, Cin, S, packBuf, outp, Cout, Cout, mk, mk,
                scaleBuf, shiftBuf, resid, S, 0);
        else
            k_conv<MODE_CONV3, false><<<g, 32, 0, stream>>>(
                in, Cin, S, packBuf, outp, Cout, Cout, mk, mk,
                nullptr, nullptr, resid, S, 0);
    };

    // ---- build grid / masks ----
    hipMemsetAsync(grid0,   0, Vl[0] * sizeof(float), stream);
    hipMemsetAsync(mask[0], 0, Vl[0] * sizeof(float), stream);
    hipMemsetAsync(ncnt,    0, 8 * sizeof(float), stream);
    k_scatter<<<(N + 255) / 256, 256, 0, stream>>>(coords, features, N, grid0, mask[0]);
    for (int l = 1; l < 5; ++l)
        k_maxpool<<<(unsigned)((Vl[l] + 255) / 256), 256, 0, stream>>>(mask[l - 1], Sl[l], mask[l]);
    for (int l = 0; l < 5; ++l)
        k_count<<<64, 256, 0, stream>>>(mask[l], (int)Vl[l], ncnt + l);

    // conv_in: 3^3, 1 -> 16 (raw input, no BN)
    conv3(grid0, 1, 96, conv_in_w, x[0], 16, mask[0], false, nullptr);

    // ---- encoder (pre blocks + down) ----
    for (int l = 0; l < 5; ++l) {
        const int p = pl[l];
        const float* w1s[2] = { Lw[l].q0w1, Lw[l].q1w1 };
        const float* w2s[2] = { Lw[l].q0w2, Lw[l].q1w2 };
        for (int b = 0; b < 2; ++b) {
            bnprep(x[l], Vl[l], p, l);
            conv3(x[l], p, Sl[l], w1s[b], T2, p, mask[l], true, nullptr);
            bnprep(T2, Vl[l], p, l);
            conv3(T2, p, Sl[l], w2s[b], x[l], p, mask[l], true, x[l]);   // x += y
        }
        if (l < 4) {
            const int pn = pl[l + 1];
            bnprep(x[l], Vl[l], p, l);
            pack(Lw[l].down, 8 * p, pn, packBuf);
            dim3 g((unsigned)((Vl[l + 1] + 15) / 16), (unsigned)((pn + 15) / 16));
            k_conv<MODE_DOWN, true><<<g, 32, 0, stream>>>(
                x[l], p, Sl[l], packBuf, x[l + 1], pn, pn, mask[l + 1], mask[l],
                scaleBuf, shiftBuf, nullptr, Sl[l + 1], 0);
        }
    }

    // ---- decoder (up + concat + post blocks) ----
    for (int l = 3; l >= 0; --l) {
        const int p = pl[l], pn = pl[l + 1];
        // up-conv: fused bnrelu(x[l+1]) -> transpose conv into CC[:, p:2p]
        bnprep(x[l + 1], Vl[l + 1], pn, l + 1);
        {
            const int nCh = (pn + 31) / 32, nG = (p + 15) / 16;
            const long perTap = (long)nCh * nG * 512;
            for (int z = 0; z < 8; ++z)   // output parity z uses weight tap 7-z
                pack(Lw[l].up + (long)(7 - z) * pn * p, pn, p, packBuf + z * perTap);
            dim3 g((unsigned)((Vl[l + 1] + 15) / 16), (unsigned)nG, 8);
            k_conv<MODE_UP, true><<<g, 32, 0, stream>>>(
                x[l + 1], pn, Sl[l + 1], packBuf, CC + p, p, 2 * p, mask[l], mask[l + 1],
                scaleBuf, shiftBuf, nullptr, Sl[l], (int)perTap);
        }
        // CC[:, :p] = x[l]
        {
            const long tot = Vl[l] * p;
            k_copych<<<(unsigned)((tot + 255) / 256), 256, 0, stream>>>(
                x[l], CC, (unsigned)Vl[l], p);
        }
        // post[0]: x = nin(CC) + conv(bnrelu(conv(bnrelu(CC))))
        bnprep(CC, Vl[l], 2 * p, l);
        conv3(CC, 2 * p, Sl[l], Lw[l].p0w1, T2, p, mask[l], true, nullptr);
        {   // NIN reads raw CC (no BN)
            pack(Lw[l].nin, 2 * p, p, packBuf);
            dim3 g((unsigned)((Vl[l] + 15) / 16), (unsigned)((p + 15) / 16));
            k_conv<MODE_NIN, false><<<g, 32, 0, stream>>>(
                CC, 2 * p, Sl[l], packBuf, x[l], p, p, mask[l], mask[l],
                nullptr, nullptr, nullptr, Sl[l], 0);
        }
        bnprep(T2, Vl[l], p, l);
        conv3(T2, p, Sl[l], Lw[l].p0w2, x[l], p, mask[l], true, x[l]);
        // post[1] (plain residual block)
        bnprep(x[l], Vl[l], p, l);
        conv3(x[l], p, Sl[l], Lw[l].p1w1, T2, p, mask[l], true, nullptr);
        bnprep(T2, Vl[l], p, l);
        conv3(T2, p, Sl[l], Lw[l].p1w2, x[l], p, mask[l], true, x[l]);
    }

    // ---- final BN (fused) + gather + linear head ----
    bnprep(x[0], Vl[0], 16, 0);
    k_gather_lin<<<(N + 255) / 256, 256, 0, stream>>>(
        x[0], coords, N, scaleBuf, shiftBuf, lin_w, lin_b, (float*)d_out);
}
